// HybridEncoder_35991825940629
// MI455X (gfx1250) — matrix operand order repaired and checked by reading, not compiled, not run on hardware
//
#include <hip/hip_runtime.h>
#include <math.h>

// ---- dims (fixed by the reference) ----
constexpr int NB  = 8;      // batch
constexpr int NN  = 1024;   // nodes / seq
constexpr int ND  = 256;    // model dim
constexpr int NH  = 8;      // heads
constexpr int NDH = 32;     // head dim
constexpr int NFF = 1024;   // ffn dim
constexpr int LG  = 2;
constexpr int LT  = 2;
constexpr int MROWS = NB * NN;          // 8192
constexpr int HGC   = NH * ND;          // 2048 (GAT per-head concat width)

typedef __attribute__((ext_vector_type(16))) _Float16 v16h;
typedef __attribute__((ext_vector_type(8)))  float    v8f;
typedef __attribute__((ext_vector_type(4)))  _Float16 h4;   // 8-byte packed f16 quad

union AF { unsigned int u[8]; v16h v; };
union KV8 { uint4 q; uint2 d[2]; _Float16 h[8]; };

__device__ __forceinline__ v8f wmma_f16(const AF& a, const AF& b, v8f c) {
  return __builtin_amdgcn_wmma_f32_16x16x32_f16(
      /*neg_a=*/false, a.v, /*neg_b=*/false, b.v,
      /*c_mod=*/(short)0, c, /*reuse_a=*/false, /*reuse_b=*/false);
}

// A-fragment K-offset pattern for 16x32 f16 A matrix (per ISA table)
__device__ __forceinline__ int a_koff(int g, int lh) {
  return ((g >> 2) << 4) + (lh << 3) + ((g & 3) << 1);
}

__device__ __forceinline__ h4 cvt_h4(float4 f) {
  h4 o; o.x = (_Float16)f.x; o.y = (_Float16)f.y;
        o.z = (_Float16)f.z; o.w = (_Float16)f.w;
  return o;
}

// =====================================================================
// Adjacency bit-pack: adjT[b][t][w] bit i = (mask[b][s][t]>0 || s==t), s=w*32+i
// =====================================================================
__global__ void pack_adj_kernel(const int* __restrict__ mask,
                                unsigned int* __restrict__ adjT) {
  int idx = blockIdx.x * blockDim.x + threadIdx.x;   // NB*NN*32
  if (idx >= NB * NN * 32) return;
  int wi = idx & 31;
  int t  = (idx >> 5) & (NN - 1);
  int b  = idx >> 15;
  unsigned int w = 0;
  for (int bit = 0; bit < 32; ++bit) {
    int s = wi * 32 + bit;
    int mv = mask[((size_t)(b * NN) + s) * NN + t];
    if (mv > 0 || s == t) w |= (1u << bit);
  }
  adjT[((size_t)(b * NN) + t) * 32 + wi] = w;
}

// =====================================================================
// Simple weight transpose: dst[c][r] = src[r][c]
// =====================================================================
__global__ void transpose_kernel(const float* __restrict__ src,
                                 float* __restrict__ dst, int R, int C) {
  int idx = blockIdx.x * blockDim.x + threadIdx.x;
  if (idx >= R * C) return;
  int r = idx / C, c = idx % C;
  dst[(size_t)c * R + r] = src[idx];
}

// =====================================================================
// Generic WMMA GEMM:  C[M,N] = A[M,K] (f32) * BT[N,K]^T (f32) + bias
// block = 128 threads (4 waves), block tile 64(M) x 64(N), K step 32
// =====================================================================
__global__ __launch_bounds__(128) void gemm_bt_kernel(
    const float* __restrict__ A, const float* __restrict__ BT,
    const float* __restrict__ bias, float* __restrict__ Cf,
    _Float16* __restrict__ Ch, int M, int N, int K, int actRelu) {
  __shared__ _Float16 As[64][40];
  __shared__ _Float16 Bs[64][40];
  const int tid = threadIdx.x;
  const int wave = tid >> 5, lane = tid & 31;
  const int lh = lane >> 4, lm = lane & 15;
  const int m0 = blockIdx.y * 64, n0 = blockIdx.x * 64;

  v8f acc[4];
  for (int i = 0; i < 4; ++i)
    for (int j = 0; j < 8; ++j) acc[i][j] = 0.f;

  for (int k0 = 0; k0 < K; k0 += 32) {
    // --- batched staging: issue all 8 global b128 loads, then convert ---
    float4 fa[4], fb[4];
#pragma unroll
    for (int j = 0; j < 4; ++j) {
      int idx = tid + j * 128;
      int r = idx >> 3, c4 = (idx & 7) << 2;
      fa[j] = *(const float4*)&A[(size_t)(m0 + r) * K + k0 + c4];
      fb[j] = *(const float4*)&BT[(size_t)(n0 + r) * K + k0 + c4];
    }
#pragma unroll
    for (int j = 0; j < 4; ++j) {
      int idx = tid + j * 128;
      int r = idx >> 3, c4 = (idx & 7) << 2;
      *(h4*)&As[r][c4] = cvt_h4(fa[j]);
      *(h4*)&Bs[r][c4] = cvt_h4(fb[j]);
    }
    if (k0 + 32 < K) {
      __builtin_prefetch(&A[(size_t)(m0 + (tid >> 1)) * K + k0 + 32], 0, 3);
      __builtin_prefetch(&BT[(size_t)(n0 + (tid >> 1)) * K + k0 + 32], 0, 3);
    }
    __syncthreads();

    // --- batched fragment loads, then back-to-back WMMAs ---
    AF a;
#pragma unroll
    for (int g = 0; g < 8; ++g)
      a.u[g] = *(const unsigned int*)&As[wave * 16 + lm][a_koff(g, lh)];
    AF b[4];
#pragma unroll
    for (int ns = 0; ns < 4; ++ns)
#pragma unroll
      for (int g = 0; g < 8; ++g)
        b[ns].u[g] = *(const unsigned int*)&Bs[ns * 16 + lm][a_koff(g, lh)];
#pragma unroll
    for (int ns = 0; ns < 4; ++ns)
      acc[ns] = wmma_f16(a, b[ns], acc[ns]);
    __syncthreads();
  }

#pragma unroll
  for (int ns = 0; ns < 4; ++ns) {
    int n = n0 + ns * 16 + lm;
    float bv = bias ? bias[n] : 0.f;
#pragma unroll
    for (int r = 0; r < 8; ++r) {
      int m = m0 + wave * 16 + lh * 8 + r;
      float v = acc[ns][r] + bv;
      if (actRelu) v = fmaxf(v, 0.f);
      if (Ch) Ch[(size_t)m * N + n] = (_Float16)v;
      else    Cf[(size_t)m * N + n] = v;
    }
  }
}

// =====================================================================
// GAT attention coefficients: a_src/a_dst[bh][n] = <h[b,n,h,:], att[h,:]>
// one wave32 per (b,n,h)
// =====================================================================
__global__ __launch_bounds__(256) void gat_coef_kernel(
    const float* __restrict__ hgat, const float* __restrict__ attS,
    const float* __restrict__ attD, float* __restrict__ a_src,
    float* __restrict__ a_dst) {
  int wid = blockIdx.x * 8 + (threadIdx.x >> 5);
  int lane = threadIdx.x & 31;
  if (wid >= NB * NN * NH) return;
  int h = wid % NH;
  int n = (wid / NH) % NN;
  int b = wid / (NH * NN);
  const float* row = hgat + ((size_t)(b * NN) + n) * HGC + h * ND;
  const float* as = attS + h * ND;
  const float* ad = attD + h * ND;
  float s = 0.f, d = 0.f;
  for (int c = lane; c < ND; c += 32) {
    float hv = row[c];
    s += hv * as[c];
    d += hv * ad[c];
  }
  for (int m = 16; m >= 1; m >>= 1) {
    s += __shfl_xor(s, m);
    d += __shfl_xor(d, m);
  }
  if (lane == 0) {
    int bh = b * NH + h;
    a_src[(size_t)bh * NN + n] = s;
    a_dst[(size_t)bh * NN + n] = d;
  }
}

// =====================================================================
// GAT softmax row stats (online max/sum over masked rank-1 scores)
// one thread per (b,h,t)
// =====================================================================
__global__ void gat_stats_kernel(const float* __restrict__ a_src,
                                 const float* __restrict__ a_dst,
                                 const unsigned int* __restrict__ adjT,
                                 float* __restrict__ rmax,
                                 float* __restrict__ rsinv) {
  int idx = blockIdx.x * blockDim.x + threadIdx.x;   // NB*NH*NN
  if (idx >= NB * NH * NN) return;
  int t = idx % NN;
  int bh = idx / NN;
  int b = bh / NH;
  float ad = a_dst[(size_t)bh * NN + t];
  float m = -1e30f, sum = 0.f;
  for (int sw = 0; sw < NN / 32; ++sw) {
    unsigned int w = adjT[((size_t)(b * NN) + t) * 32 + sw];
    if (!w) continue;
    for (int bit = 0; bit < 32; ++bit) {
      if ((w >> bit) & 1u) {
        float e = ad + a_src[(size_t)bh * NN + sw * 32 + bit];
        e = e > 0.f ? e : 0.2f * e;              // leaky_relu(0.2)
        if (e > m) { sum = sum * __expf(m - e) + 1.f; m = e; }
        else       { sum += __expf(e - m); }
      }
    }
  }
  rmax[idx] = m;
  rsinv[idx] = 1.f / sum;
}

// =====================================================================
// GAT aggregation: out[b,t,c] = (1/H) * sum_h sum_s alpha[bh,t,s]*H[b,s,h,c] + bias[c]
// block tile 64(t) x 64(c), accumulates over heads + s-tiles via WMMA
// =====================================================================
__global__ __launch_bounds__(128) void gat_agg_kernel(
    const float* __restrict__ hgat, const float* __restrict__ a_src,
    const float* __restrict__ a_dst, const float* __restrict__ rmax,
    const float* __restrict__ rsinv, const unsigned int* __restrict__ adjT,
    const float* __restrict__ gbias, float* __restrict__ out) {
  __shared__ _Float16 Al[64][40];   // alpha tile [t][s]
  __shared__ _Float16 Bl[64][40];   // H^T tile   [c][s]
  const int tid = threadIdx.x;
  const int wave = tid >> 5, lane = tid & 31;
  const int lh = lane >> 4, lm = lane & 15;
  const int c0 = blockIdx.x * 64, t0 = blockIdx.y * 64, b = blockIdx.z;

  v8f acc[4];
  for (int i = 0; i < 4; ++i)
    for (int j = 0; j < 8; ++j) acc[i][j] = 0.f;

  for (int h = 0; h < NH; ++h) {
    int bh = b * NH + h;
    for (int st = 0; st < NN; st += 32) {
      // H^T tile: batched float4 loads, transposed stores
      float4 fh[4];
#pragma unroll
      for (int j = 0; j < 4; ++j) {
        int idx = tid + j * 128;          // 512 float4 chunks
        int s = idx >> 4, cq = (idx & 15) << 2;
        fh[j] = *(const float4*)&hgat[((size_t)(b * NN) + st + s) * HGC +
                                      h * ND + c0 + cq];
      }
#pragma unroll
      for (int j = 0; j < 4; ++j) {
        int idx = tid + j * 128;
        int s = idx >> 4, cq = (idx & 15) << 2;
        Bl[cq][s]     = (_Float16)fh[j].x;
        Bl[cq + 1][s] = (_Float16)fh[j].y;
        Bl[cq + 2][s] = (_Float16)fh[j].z;
        Bl[cq + 3][s] = (_Float16)fh[j].w;
      }
      // alpha tile: each thread owns one source column s, 16 target rows
      {
        int s = tid & 31;
        int tb = tid >> 5;
        float asv = a_src[(size_t)bh * NN + st + s];
        unsigned int bitm = 1u << s;
#pragma unroll
        for (int k2 = 0; k2 < 16; ++k2) {
          int t = tb + k2 * 4;
          int tg = t0 + t;
          unsigned int w = adjT[((size_t)(b * NN) + tg) * 32 + (st >> 5)];
          float p = 0.f;
          if (w & bitm) {
            float e = a_dst[(size_t)bh * NN + tg] + asv;
            e = e > 0.f ? e : 0.2f * e;
            p = __expf(e - rmax[(size_t)bh * NN + tg]) *
                rsinv[(size_t)bh * NN + tg];
          }
          Al[t][s] = (_Float16)p;
        }
      }
      __syncthreads();

      AF a;
#pragma unroll
      for (int g = 0; g < 8; ++g)
        a.u[g] = *(const unsigned int*)&Al[wave * 16 + lm][a_koff(g, lh)];
      AF bb[4];
#pragma unroll
      for (int ns = 0; ns < 4; ++ns)
#pragma unroll
        for (int g = 0; g < 8; ++g)
          bb[ns].u[g] =
              *(const unsigned int*)&Bl[ns * 16 + lm][a_koff(g, lh)];
#pragma unroll
      for (int ns = 0; ns < 4; ++ns)
        acc[ns] = wmma_f16(a, bb[ns], acc[ns]);
      __syncthreads();
    }
  }

#pragma unroll
  for (int ns = 0; ns < 4; ++ns) {
    int c = c0 + ns * 16 + lm;
#pragma unroll
    for (int r = 0; r < 8; ++r) {
      int tg = t0 + wave * 16 + lh * 8 + r;
      out[((size_t)(b * NN) + tg) * ND + c] = acc[ns][r] * (1.f / NH) + gbias[c];
    }
  }
}

// =====================================================================
// Masked multi-head flash attention: o = softmax(QK^T * scale, mask) V
// grid (N/64, H, B); block = 4 waves; q/k/v in f16 [B*N][256]
// =====================================================================
__global__ __launch_bounds__(128) void attn_kernel(
    const _Float16* __restrict__ q, const _Float16* __restrict__ k,
    const _Float16* __restrict__ v, const unsigned int* __restrict__ adjT,
    float* __restrict__ o) {
  __shared__ _Float16 Ks[32][36];        // [s][d]
  __shared__ _Float16 VT[32][36];        // [d][s]
  __shared__ _Float16 Ps[4][16][36];     // per-wave P tile [m][s]
  const int tid = threadIdx.x;
  const int wave = tid >> 5, lane = tid & 31;
  const int lh = lane >> 4, lm = lane & 15;
  const int t0 = blockIdx.x * 64, h = blockIdx.y, b = blockIdx.z;
  const float scale = 0.1767766952966369f;   // 1/sqrt(32)

  AF aq;
  {
    const size_t rowQ = ((size_t)(b * NN) + t0 + wave * 16 + lm) * ND + h * NDH;
#pragma unroll
    for (int g = 0; g < 8; ++g)
      aq.u[g] = *(const unsigned int*)&q[rowQ + a_koff(g, lh)];
  }

  float rmax[8], rsum[8];
  v8f acc[2];
#pragma unroll
  for (int r = 0; r < 8; ++r) {
    rmax[r] = -1e30f; rsum[r] = 0.f;
    acc[0][r] = 0.f; acc[1][r] = 0.f;
  }
  v8f zc;
  for (int r = 0; r < 8; ++r) zc[r] = 0.f;

  for (int st = 0; st < NN; st += 32) {
    // vectorized K/V staging: both 16-byte loads in flight before use
    {
      int s = tid >> 2, d8 = (tid & 3) << 3;
      size_t gi = ((size_t)(b * NN) + st + s) * ND + h * NDH + d8;
      KV8 kv, vv;
      kv.q = *(const uint4*)&k[gi];
      vv.q = *(const uint4*)&v[gi];
      *(uint2*)&Ks[s][d8]     = kv.d[0];
      *(uint2*)&Ks[s][d8 + 4] = kv.d[1];
#pragma unroll
      for (int e = 0; e < 8; ++e) VT[d8 + e][s] = vv.h[e];
    }
    __syncthreads();

    // scores: two 16x16 tiles covering s-columns [st, st+32)
    AF bk0, bk1;
#pragma unroll
    for (int g = 0; g < 8; ++g) {
      bk0.u[g] = *(const unsigned int*)&Ks[lm][a_koff(g, lh)];
      bk1.u[g] = *(const unsigned int*)&Ks[16 + lm][a_koff(g, lh)];
    }
    v8f sc0 = wmma_f16(aq, bk0, zc);
    v8f sc1 = wmma_f16(aq, bk1, zc);

    // online softmax update (row = 8*lh + r; 16-lane row reductions)
#pragma unroll
    for (int r = 0; r < 8; ++r) {
      int t = t0 + wave * 16 + lh * 8 + r;
      unsigned int w = adjT[((size_t)(b * NN) + t) * 32 + (st >> 5)];
      bool al0 = (w >> lm) & 1u;
      bool al1 = (w >> (16 + lm)) & 1u;
      float v0 = al0 ? sc0[r] * scale : -1e30f;
      float v1 = al1 ? sc1[r] * scale : -1e30f;
      float mx = fmaxf(v0, v1);
      mx = fmaxf(mx, __shfl_xor(mx, 1));
      mx = fmaxf(mx, __shfl_xor(mx, 2));
      mx = fmaxf(mx, __shfl_xor(mx, 4));
      mx = fmaxf(mx, __shfl_xor(mx, 8));
      float nm = fmaxf(rmax[r], mx);
      float sf = __expf(rmax[r] - nm);
      float p0 = al0 ? __expf(v0 - nm) : 0.f;
      float p1 = al1 ? __expf(v1 - nm) : 0.f;
      float ps = p0 + p1;
      ps += __shfl_xor(ps, 1);
      ps += __shfl_xor(ps, 2);
      ps += __shfl_xor(ps, 4);
      ps += __shfl_xor(ps, 8);
      rsum[r] = rsum[r] * sf + ps;
      rmax[r] = nm;
      acc[0][r] *= sf;
      acc[1][r] *= sf;
      Ps[wave][lh * 8 + r][lm] = (_Float16)p0;
      Ps[wave][lh * 8 + r][16 + lm] = (_Float16)p1;
    }
    // intra-wave LDS RAW: CDNA5 split-counter wait
    asm volatile("s_wait_dscnt 0" ::: "memory");

    // P @ V: batch both fragment loads, then both WMMAs
    AF ap, bv0, bv1;
#pragma unroll
    for (int g = 0; g < 8; ++g) {
      ap.u[g]  = *(const unsigned int*)&Ps[wave][lm][a_koff(g, lh)];
      bv0.u[g] = *(const unsigned int*)&VT[lm][a_koff(g, lh)];
      bv1.u[g] = *(const unsigned int*)&VT[16 + lm][a_koff(g, lh)];
    }
    acc[0] = wmma_f16(ap, bv0, acc[0]);
    acc[1] = wmma_f16(ap, bv1, acc[1]);
    __syncthreads();
  }

#pragma unroll
  for (int ds = 0; ds < 2; ++ds) {
#pragma unroll
    for (int r = 0; r < 8; ++r) {
      int t = t0 + wave * 16 + lh * 8 + r;
      o[((size_t)(b * NN) + t) * ND + h * NDH + ds * 16 + lm] =
          acc[ds][r] / rsum[r];
    }
  }
}

// =====================================================================
// y = LayerNorm(a + (relu?)(b)) * s + bi    (rows of 256, one block/row)
// =====================================================================
__global__ __launch_bounds__(256) void add_ln_kernel(
    const float* __restrict__ a, const float* __restrict__ bm, int reluB,
    const float* __restrict__ s, const float* __restrict__ bi,
    float* __restrict__ y) {
  __shared__ float red[256];
  int row = blockIdx.x, i = threadIdx.x;
  float bv = bm[(size_t)row * ND + i];
  if (reluB) bv = fmaxf(bv, 0.f);
  float v = a[(size_t)row * ND + i] + bv;
  red[i] = v;
  __syncthreads();
  for (int off = 128; off > 0; off >>= 1) {
    if (i < off) red[i] += red[i + off];
    __syncthreads();
  }
  float mean = red[0] * (1.f / ND);
  __syncthreads();
  float dv = v - mean;
  red[i] = dv * dv;
  __syncthreads();
  for (int off = 128; off > 0; off >>= 1) {
    if (i < off) red[i] += red[i + off];
    __syncthreads();
  }
  float var = red[0] * (1.f / ND);
  y[(size_t)row * ND + i] = dv * rsqrtf(var + 1e-5f) * s[i] + bi[i];
}

// =====================================================================
// host-side launcher
// =====================================================================
extern "C" void kernel_launch(void* const* d_in, const int* in_sizes, int n_in,
                              void* d_out, int out_size, void* d_ws, size_t ws_size,
                              hipStream_t stream) {
  (void)in_sizes; (void)n_in; (void)out_size; (void)ws_size;

  const float* emb      = (const float*)d_in[0];
  const int*   mask     = (const int*)d_in[2];
  const float* gat_W    = (const float*)d_in[4];
  const float* gat_aS   = (const float*)d_in[5];
  const float* gat_aD   = (const float*)d_in[6];
  const float* gat_bias = (const float*)d_in[7];
  const float* gln_s    = (const float*)d_in[8];
  const float* gln_b    = (const float*)d_in[9];
  const float* Wq = (const float*)d_in[10]; const float* bq = (const float*)d_in[11];
  const float* Wk = (const float*)d_in[12]; const float* bk = (const float*)d_in[13];
  const float* Wv = (const float*)d_in[14]; const float* bv = (const float*)d_in[15];
  const float* Wo = (const float*)d_in[16]; const float* bo = (const float*)d_in[17];
  const float* W1 = (const float*)d_in[18]; const float* b1 = (const float*)d_in[19];
  const float* W2 = (const float*)d_in[20]; const float* b2 = (const float*)d_in[21];
  const float* ln1_s = (const float*)d_in[22]; const float* ln1_b = (const float*)d_in[23];
  const float* ln2_s = (const float*)d_in[24]; const float* ln2_b = (const float*)d_in[25];

  // ---- workspace carve-up ----
  char* ws = (char*)d_ws;
  size_t off = 0;
  auto take = [&](size_t bytes) -> char* {
    char* p = ws + off;
    off += (bytes + 255) & ~(size_t)255;
    return p;
  };
  const size_t LSTRIDE = 4 * 65536 + 2 * 262144;           // per-layer transposed wts
  float*        wtAll = (float*)take(LT * LSTRIDE * 4);
  unsigned int* adjT  = (unsigned int*)take((size_t)NB * NN * 32 * 4);
  float* a_src  = (float*)take((size_t)NB * NH * NN * 4);
  float* a_dst  = (float*)take((size_t)NB * NH * NN * 4);
  float* rmaxb  = (float*)take((size_t)NB * NH * NN * 4);
  float* rsinvb = (float*)take((size_t)NB * NH * NN * 4);
  float* hgat   = (float*)take((size_t)MROWS * HGC * 4);
  float* xga    = (float*)take((size_t)MROWS * ND * 4);
  float* xgb    = (float*)take((size_t)MROWS * ND * 4);
  float* xbuf   = (float*)take((size_t)MROWS * ND * 4);
  _Float16* q16 = (_Float16*)take((size_t)MROWS * ND * 2);
  _Float16* k16 = (_Float16*)take((size_t)MROWS * ND * 2);
  _Float16* v16 = (_Float16*)take((size_t)MROWS * ND * 2);
  float* obuf   = (float*)take((size_t)MROWS * ND * 4);
  float* bufP   = (float*)take((size_t)MROWS * ND * 4);
  float* midb   = (float*)take((size_t)MROWS * NFF * 4);

  // ---- adjacency pack ----
  pack_adj_kernel<<<(NB * NN * 32 + 255) / 256, 256, 0, stream>>>(mask, adjT);

  // ---- transpose transformer weights ----
  for (int l = 0; l < LT; ++l) {
    float* base = wtAll + l * LSTRIDE;
    transpose_kernel<<<(65536 + 255) / 256, 256, 0, stream>>>(Wq + l * 65536, base + 0,      ND, ND);
    transpose_kernel<<<(65536 + 255) / 256, 256, 0, stream>>>(Wk + l * 65536, base + 65536,  ND, ND);
    transpose_kernel<<<(65536 + 255) / 256, 256, 0, stream>>>(Wv + l * 65536, base + 131072, ND, ND);
    transpose_kernel<<<(65536 + 255) / 256, 256, 0, stream>>>(Wo + l * 65536, base + 196608, ND, ND);
    transpose_kernel<<<(262144 + 255) / 256, 256, 0, stream>>>(W1 + l * 262144, base + 262144, ND, NFF);
    transpose_kernel<<<(262144 + 255) / 256, 256, 0, stream>>>(W2 + l * 262144, base + 524288, NFF, ND);
  }

  // ---- GAT stack ----
  const float* xin = emb;
  float* gouts[2] = {xga, xgb};
  for (int l = 0; l < LG; ++l) {
    // h = x @ W_h^T for all heads  (gat_W[l] is already [H*C][D] = B^T)
    gemm_bt_kernel<<<dim3(HGC / 64, MROWS / 64), 128, 0, stream>>>(
        xin, gat_W + (size_t)l * NH * ND * ND, nullptr, hgat, nullptr,
        MROWS, HGC, ND, 0);
    gat_coef_kernel<<<(NB * NN * NH) / 8, 256, 0, stream>>>(
        hgat, gat_aS + l * NH * ND, gat_aD + l * NH * ND, a_src, a_dst);
    gat_stats_kernel<<<(NB * NH * NN + 255) / 256, 256, 0, stream>>>(
        a_src, a_dst, adjT, rmaxb, rsinvb);
    gat_agg_kernel<<<dim3(ND / 64, NN / 64, NB), 128, 0, stream>>>(
        hgat, a_src, a_dst, rmaxb, rsinvb, adjT, gat_bias + l * ND, gouts[l]);
    xin = gouts[l];
  }
  add_ln_kernel<<<MROWS, 256, 0, stream>>>(emb, xin, 1, gln_s, gln_b, xbuf);

  // ---- transformer encoder layers ----
  for (int l = 0; l < LT; ++l) {
    float* base = wtAll + l * LSTRIDE;
    gemm_bt_kernel<<<dim3(ND / 64, MROWS / 64), 128, 0, stream>>>(
        xbuf, base + 0, bq + l * ND, nullptr, q16, MROWS, ND, ND, 0);
    gemm_bt_kernel<<<dim3(ND / 64, MROWS / 64), 128, 0, stream>>>(
        xbuf, base + 65536, bk + l * ND, nullptr, k16, MROWS, ND, ND, 0);
    gemm_bt_kernel<<<dim3(ND / 64, MROWS / 64), 128, 0, stream>>>(
        xbuf, base + 131072, bv + l * ND, nullptr, v16, MROWS, ND, ND, 0);

    attn_kernel<<<dim3(NN / 64, NH, NB), 128, 0, stream>>>(q16, k16, v16, adjT, obuf);

    gemm_bt_kernel<<<dim3(ND / 64, MROWS / 64), 128, 0, stream>>>(
        obuf, base + 196608, bo + l * ND, bufP, nullptr, MROWS, ND, ND, 0);
    add_ln_kernel<<<MROWS, 256, 0, stream>>>(xbuf, bufP, 0,
                                             ln1_s + l * ND, ln1_b + l * ND, xbuf);

    gemm_bt_kernel<<<dim3(NFF / 64, MROWS / 64), 128, 0, stream>>>(
        xbuf, base + 262144, b1 + l * NFF, midb, nullptr, MROWS, NFF, ND, 1);
    gemm_bt_kernel<<<dim3(ND / 64, MROWS / 64), 128, 0, stream>>>(
        midb, base + 524288, b2 + l * ND, bufP, nullptr, MROWS, ND, NFF, 0);

    float* yout = (l == LT - 1) ? (float*)d_out : xbuf;
    add_ln_kernel<<<MROWS, 256, 0, stream>>>(xbuf, bufP, 0,
                                             ln2_s + l * ND, ln2_b + l * ND, yout);
  }
}
